// DMPNN_46720654246230
// MI455X (gfx1250) — matrix operand order repaired and checked by reading, not compiled, not run on hardware
//
#include <hip/hip_runtime.h>
#include <hip/hip_bf16.h>

typedef __attribute__((ext_vector_type(16))) _Float16 v16h;
typedef __attribute__((ext_vector_type(8)))  _Float16 v8h;
typedef __attribute__((ext_vector_type(8)))  float    v8f;

#define N_ATOMS_C 98304
#define N_EDGES_C 204800
#define N_MOLS_C  4096
#define APM_C     24
#define AF_DIM_C  73
#define BF_DIM_C  13
#define HID_C     300
#define HP_C      320      // padded hidden (both N and K pad of 300)
#define NT_C      20       // 320/16 N tiles
#define DEPTH_C   3
#define KC_MAX_C  12       // largest K-chunk count (atom readout GEMM)

// ---------------------------------------------------------------------------
// Weight prep: W [outDim, inDim] f32 row-major  ->  f16 fragment-ordered
// layout [NT][KC][512] where within a 512-half tile:
//   idx = lane*16 + h,  N = nt*16 + (lane&15),  K = kc*32 + (lane<16?0:8) + (h<8 ? h : 8+h)
// This matches the CDNA5 16-bit B-matrix (32x16) per-lane VGPR layout, so the
// GEMM reads each B fragment as one contiguous 32B chunk per lane.
// ---------------------------------------------------------------------------
__global__ void prep_wfrag(const float* __restrict__ W, _Float16* __restrict__ out,
                           int outDim, int inDim, int KC) {
    int idx = blockIdx.x * blockDim.x + threadIdx.x;
    int total = NT_C * KC * 512;
    if (idx >= total) return;
    int h    = idx & 15;
    int lane = (idx >> 4) & 31;
    int tile = idx >> 9;             // nt*KC + kc
    int kc = tile % KC;
    int nt = tile / KC;
    int n = nt * 16 + (lane & 15);
    int k = kc * 32 + ((lane < 16) ? 0 : 8) + ((h < 8) ? h : 8 + h);
    float v = 0.0f;
    if (n < outDim && k < inDim) v = W[n * inDim + k];
    out[idx] = (_Float16)v;
}

__global__ void pad_bias(const float* __restrict__ b, float* __restrict__ out) {
    int t = blockIdx.x * blockDim.x + threadIdx.x;
    if (t < HP_C) out[t] = (t < HID_C) ? b[t] : 0.0f;
}

// edge_in = concat(atom_feats[src], bond_feats) -> f16 [E][96] (pad 86..95 = 0)
__global__ void build_edge_in(const float* __restrict__ af, const float* __restrict__ bf,
                              const int* __restrict__ src, _Float16* __restrict__ out) {
    int t = blockIdx.x * blockDim.x + threadIdx.x;
    if (t >= N_EDGES_C * 96) return;
    int e = t / 96;
    int c = t - e * 96;
    float v = 0.0f;
    if (c < AF_DIM_C)                 v = af[(size_t)src[e] * AF_DIM_C + c];
    else if (c < AF_DIM_C + BF_DIM_C) v = bf[(size_t)e * BF_DIM_C + (c - AF_DIM_C)];
    out[t] = (_Float16)v;
}

// agg[idx[e]] += h[e]  (f32 atomics resolve in L2; 126MB agg fits in 192MB L2)
__global__ void scatter_add(const float* __restrict__ h, const int* __restrict__ idx,
                            float* __restrict__ agg) {
    int t = blockIdx.x * blockDim.x + threadIdx.x;
    if (t >= N_EDGES_C * HID_C) return;
    int e = t / HID_C;
    int c = t - e * HID_C;
    atomicAdd(&agg[(size_t)idx[e] * HP_C + c], h[(size_t)e * HP_C + c]);
}

// msg = (f16)(agg[src[e]] - h[rev[e]])   (padded cols are 0-0=0)
__global__ void build_msg(const float* __restrict__ agg, const float* __restrict__ h,
                          const int* __restrict__ src, const int* __restrict__ rev,
                          _Float16* __restrict__ msg) {
    int t = blockIdx.x * blockDim.x + threadIdx.x;
    if (t >= N_EDGES_C * HP_C) return;
    int e = t / HP_C;
    int c = t - e * HP_C;
    float v = agg[(size_t)src[e] * HP_C + c] - h[(size_t)rev[e] * HP_C + c];
    msg[t] = (_Float16)v;
}

// atom_in = concat(atom_feats, m_atom) -> f16 [A][384] (pad 373..383 = 0)
__global__ void build_atom_in(const float* __restrict__ af, const float* __restrict__ magg,
                              _Float16* __restrict__ out) {
    int t = blockIdx.x * blockDim.x + threadIdx.x;
    if (t >= N_ATOMS_C * 384) return;
    int a = t / 384;
    int c = t - a * 384;
    float v = 0.0f;
    if (c < AF_DIM_C)               v = af[(size_t)a * AF_DIM_C + c];
    else if (c < AF_DIM_C + HID_C)  v = magg[(size_t)a * HP_C + (c - AF_DIM_C)];
    out[t] = (_Float16)v;
}

// mol_vec = mean over 24 contiguous atoms -> f16 [M][HP]
__global__ void pool_mean(const float* __restrict__ hatom, _Float16* __restrict__ mv) {
    int t = blockIdx.x * blockDim.x + threadIdx.x;
    if (t >= N_MOLS_C * HP_C) return;
    int m = t / HP_C;
    int c = t - m * HP_C;
    const float* p = hatom + (size_t)m * APM_C * HP_C + c;
    float s = 0.0f;
#pragma unroll
    for (int i = 0; i < APM_C; ++i) s += p[(size_t)i * HP_C];
    mv[t] = (_Float16)(s * (1.0f / APM_C));
}

// out[m][k] = hidden[m][:300] . W2[k][:300] + b2[k]
__global__ void head_kernel(const float* __restrict__ hidden, const float* __restrict__ W2,
                            const float* __restrict__ b2, float* __restrict__ out) {
    int t = blockIdx.x * blockDim.x + threadIdx.x;
    if (t >= N_MOLS_C * 3) return;
    int m = t / 3;
    int k = t - m * 3;
    const float* hr = hidden + (size_t)m * HP_C;
    const float* w  = W2 + (size_t)k * HID_C;
    float s = b2[k];
    for (int i = 0; i < HID_C; ++i) s += hr[i] * w[i];
    out[t] = s;
}

// ---------------------------------------------------------------------------
// WMMA GEMM:  Out[rows][HP] = relu( A[rows][KP] @ Bfrag + addend + bias )
//   A: f16 row-major (KP = 32*KC), loaded as CDNA5 16-bit 16x32 A fragments
//   Bfrag: f16 fragment-ordered [NT][KC][512]
// Block = 256 threads = 8 waves. The block's 4 weight column-tiles
// ([4][KC][512] halves, <=48KB) are staged into LDS once (coalesced) and
// re-read by all 8 waves via ds loads. Each wave computes a 32x64 strip
// (2 M-tiles x 4 N-tiles = 8 accumulators, 8 WMMAs per K-chunk), with the
// next K-chunk's A fragments prefetched from global before the current
// chunk's WMMAs issue, so VMEM latency hides under the MMA pipe.
// ---------------------------------------------------------------------------
__global__ __launch_bounds__(256)
void gemm_wmma(const _Float16* __restrict__ A, const _Float16* __restrict__ Bfrag,
               const float* __restrict__ addend, const float* __restrict__ bias,
               float* __restrict__ Out, int KP, int KC) {
    __shared__ _Float16 Bs[4 * KC_MAX_C * 512];   // 48KB max (320KB/WGP LDS)

    int lane = threadIdx.x & 31;
    int wave = threadIdx.x >> 5;
    int nt0  = blockIdx.y * 4;

    // Cooperative stage: the 4 column tiles are one contiguous slab in Bfrag.
    {
        const uint4* g = (const uint4*)(Bfrag + ((size_t)(nt0 * KC) << 9));
        uint4* s = (uint4*)Bs;
        int n16 = KC * 256;                       // 16B chunks in the slab
        for (int i = threadIdx.x; i < n16; i += 256) s[i] = g[i];
    }
    __syncthreads();

    int m0 = (blockIdx.x * 8 + wave) * 32;        // 32 rows per wave
    // A fragment addressing: lane holds row (lane&15); halves 0..7 at
    // K = k0..k0+7, halves 8..15 at K = k0+16..k0+23, k0 = kc*32 + (lane<16?0:8)
    const _Float16* Ab0 = A + (size_t)(m0 + (lane & 15)) * KP + ((lane < 16) ? 0 : 8);
    const _Float16* Ab1 = Ab0 + (size_t)16 * KP;

    v8f acc[2][4];
#pragma unroll
    for (int g = 0; g < 2; ++g)
#pragma unroll
        for (int t = 0; t < 4; ++t)
            acc[g][t] = (v8f){0.f, 0.f, 0.f, 0.f, 0.f, 0.f, 0.f, 0.f};

    v8h a0lo = *(const v8h*)(Ab0);
    v8h a0hi = *(const v8h*)(Ab0 + 16);
    v8h a1lo = *(const v8h*)(Ab1);
    v8h a1hi = *(const v8h*)(Ab1 + 16);

    for (int kc = 0; kc < KC; ++kc) {
        // prefetch next chunk's A fragments (uniform branch; hides VMEM latency)
        v8h n0lo = a0lo, n0hi = a0hi, n1lo = a1lo, n1hi = a1hi;
        if (kc + 1 < KC) {
            int o = (kc + 1) * 32;
            n0lo = *(const v8h*)(Ab0 + o);
            n0hi = *(const v8h*)(Ab0 + o + 16);
            n1lo = *(const v8h*)(Ab1 + o);
            n1hi = *(const v8h*)(Ab1 + o + 16);
        }
        v16h a0 = __builtin_shufflevector(a0lo, a0hi, 0, 1, 2, 3, 4, 5, 6, 7,
                                                      8, 9, 10, 11, 12, 13, 14, 15);
        v16h a1 = __builtin_shufflevector(a1lo, a1hi, 0, 1, 2, 3, 4, 5, 6, 7,
                                                      8, 9, 10, 11, 12, 13, 14, 15);
#pragma unroll
        for (int t = 0; t < 4; ++t) {
            v16h b = *(const v16h*)(Bs + (((t * KC + kc) << 9) + lane * 16));
            acc[0][t] = __builtin_amdgcn_wmma_f32_16x16x32_f16(
                false, a0, false, b, (short)0, acc[0][t], false, false);
            acc[1][t] = __builtin_amdgcn_wmma_f32_16x16x32_f16(
                false, a1, false, b, (short)0, acc[1][t], false, false);
        }
        a0lo = n0lo; a0hi = n0hi; a1lo = n1lo; a1hi = n1hi;
    }

    // C/D layout: VGPR r -> row tile_m0 + r + (lane>=16 ? 8 : 0),
    //             col = ntile*16 + (lane&15)
    int rb = (lane < 16) ? 0 : 8;
    int cb = lane & 15;
#pragma unroll
    for (int g = 0; g < 2; ++g) {
        int rowb = m0 + g * 16 + rb;
#pragma unroll
        for (int t = 0; t < 4; ++t) {
            int col = (nt0 + t) * 16 + cb;
            float bv = bias ? bias[col] : 0.0f;
#pragma unroll
            for (int r = 0; r < 8; ++r) {
                size_t o = (size_t)(rowb + r) * HP_C + col;
                float v = acc[g][t][r] + bv;
                if (addend) v += addend[o];
                Out[o] = v > 0.0f ? v : 0.0f;
            }
        }
    }
}

// ---------------------------------------------------------------------------
extern "C" void kernel_launch(void* const* d_in, const int* in_sizes, int n_in,
                              void* d_out, int out_size, void* d_ws, size_t ws_size,
                              hipStream_t stream) {
    const float* af  = (const float*)d_in[0];
    const float* bf  = (const float*)d_in[1];
    const int*   src = (const int*)d_in[2];
    const int*   dsti= (const int*)d_in[3];
    const int*   rev = (const int*)d_in[4];
    const float* W_i = (const float*)d_in[6];
    const float* W_m = (const float*)d_in[7];
    const float* W_a = (const float*)d_in[8];
    const float* b_a = (const float*)d_in[9];
    const float* W1  = (const float*)d_in[10];
    const float* b1  = (const float*)d_in[11];
    const float* W2  = (const float*)d_in[12];
    const float* b2  = (const float*)d_in[13];

    const size_t E = N_EDGES_C, A = N_ATOMS_C, M = N_MOLS_C;
    size_t off = 0;
    auto alloc = [&](size_t bytes) -> char* {
        char* p = (char*)d_ws + off;
        off += (bytes + 255) & ~(size_t)255;
        return p;
    };
    float*    hinit = (float*)   alloc(E * HP_C * 4);   // h_init; reused as h_atom
    float*    h     = (float*)   alloc(E * HP_C * 4);
    float*    agg   = (float*)   alloc(A * HP_C * 4);
    _Float16* msg16 = (_Float16*)alloc(E * HP_C * 2);
    _Float16* ein16 = (_Float16*)alloc(E * 96 * 2);
    _Float16* ain16 = (_Float16*)alloc(A * 384 * 2);
    _Float16* mv16  = (_Float16*)alloc(M * HP_C * 2);
    float*    hidden= (float*)   alloc(M * HP_C * 4);
    _Float16* wiF   = (_Float16*)alloc((size_t)NT_C * 3  * 512 * 2);
    _Float16* wmF   = (_Float16*)alloc((size_t)3 * NT_C * 10 * 512 * 2);
    _Float16* waF   = (_Float16*)alloc((size_t)NT_C * 12 * 512 * 2);
    _Float16* w1F   = (_Float16*)alloc((size_t)NT_C * 10 * 512 * 2);
    float*    baP   = (float*)   alloc(HP_C * 4);
    float*    b1P   = (float*)   alloc(HP_C * 4);

    const int TB = 256;
    auto grd = [](size_t n) { return (unsigned)((n + 255) / 256); };

    // --- weight / bias prep (small, once per launch) ---
    prep_wfrag<<<grd((size_t)NT_C * 3 * 512),  TB, 0, stream>>>(W_i, wiF, HID_C, AF_DIM_C + BF_DIM_C, 3);
    for (int d = 0; d < DEPTH_C; ++d)
        prep_wfrag<<<grd((size_t)NT_C * 10 * 512), TB, 0, stream>>>(
            W_m + (size_t)d * HID_C * HID_C, wmF + (size_t)d * NT_C * 10 * 512, HID_C, HID_C, 10);
    prep_wfrag<<<grd((size_t)NT_C * 12 * 512), TB, 0, stream>>>(W_a, waF, HID_C, AF_DIM_C + HID_C, 12);
    prep_wfrag<<<grd((size_t)NT_C * 10 * 512), TB, 0, stream>>>(W1, w1F, HID_C, HID_C, 10);
    pad_bias<<<grd(HP_C), TB, 0, stream>>>(b_a, baP);
    pad_bias<<<grd(HP_C), TB, 0, stream>>>(b1, b1P);

    // --- h_init = relu(edge_in @ W_i^T) ---
    build_edge_in<<<grd(E * 96), TB, 0, stream>>>(af, bf, src, ein16);
    gemm_wmma<<<dim3((unsigned)(E / 256), 5), TB, 0, stream>>>(
        ein16, wiF, nullptr, nullptr, hinit, 96, 3);

    // --- depth x directed message passing ---
    for (int d = 0; d < DEPTH_C; ++d) {
        const float* hcur = (d == 0) ? hinit : h;
        hipMemsetAsync(agg, 0, A * HP_C * 4, stream);
        scatter_add<<<grd(E * HID_C), TB, 0, stream>>>(hcur, src, agg);
        build_msg<<<grd(E * HP_C), TB, 0, stream>>>(agg, hcur, src, rev, msg16);
        gemm_wmma<<<dim3((unsigned)(E / 256), 5), TB, 0, stream>>>(
            msg16, wmF + (size_t)d * NT_C * 10 * 512, hinit, nullptr, h, HP_C, 10);
    }

    // --- atom readout: h_atom = relu([af ; segsum(h,dst)] @ W_a^T + b_a) ---
    hipMemsetAsync(agg, 0, A * HP_C * 4, stream);
    scatter_add<<<grd(E * HID_C), TB, 0, stream>>>(h, dsti, agg);
    build_atom_in<<<grd(A * 384), TB, 0, stream>>>(af, agg, ain16);
    gemm_wmma<<<dim3((unsigned)(A / 256), 5), TB, 0, stream>>>(
        ain16, waF, nullptr, baP, hinit /* = h_atom */, 384, 12);

    // --- per-molecule mean pool + FFN head ---
    pool_mean<<<grd(M * HP_C), TB, 0, stream>>>(hinit, mv16);
    gemm_wmma<<<dim3((unsigned)(M / 256), 5), TB, 0, stream>>>(
        mv16, w1F, nullptr, b1P, hidden, HP_C, 10);
    head_kernel<<<grd(M * 3), TB, 0, stream>>>(hidden, W2, b2, (float*)d_out);
}
